// RNNLanguageModel_7404523619241
// MI455X (gfx1250) — compile-verified, matrix operand order checked by reading, not run on hardware
//
#include <hip/hip_runtime.h>
#include <math.h>
#include <stdint.h>

// ---------------------------------------------------------------------------
// Model dimensions (compile-time constants from the reference)
// ---------------------------------------------------------------------------
#define Bsz 4
#define Tsz 1024
#define Esz 512
#define Hsz 1024
#define Ksz 512
#define Vdz 512
#define Vsz 32000

typedef __bf16 bf16_t;
typedef __attribute__((ext_vector_type(16))) __bf16 v16bf;
typedef __attribute__((ext_vector_type(8)))  float  v8f;

__device__ __forceinline__ bf16_t f2bf(float f) {
  union { float f; unsigned u; } x; x.f = f;
  unsigned r = x.u + 0x7FFFu + ((x.u >> 16) & 1u);   // round-to-nearest-even
  unsigned short h = (unsigned short)(r >> 16);
  bf16_t b;
  __builtin_memcpy(&b, &h, sizeof(h));
  return b;
}

// CDNA5 async global->LDS copy (ASYNCcnt-tracked, no VGPR roundtrip).
// LDS destination is a byte offset: low 32 bits of a generic pointer into LDS
// are the LDS offset (aperture occupies addr[63:32] per the ISA).
__device__ __forceinline__ void async_copy_b128(uint32_t lds_off, const void* gsrc) {
  asm volatile("global_load_async_to_lds_b128 %0, %1, off"
               :: "v"(lds_off), "v"((unsigned long long)(uintptr_t)gsrc)
               : "memory");
}
__device__ __forceinline__ void wait_async_le4() {
  asm volatile("s_wait_asynccnt 4" ::: "memory");
}
__device__ __forceinline__ void wait_async_0() {
  asm volatile("s_wait_asynccnt 0" ::: "memory");
}

#define WMMA_BF16(a, b, c) \
  __builtin_amdgcn_wmma_f32_16x16x32_bf16(false, (a), false, (b), (short)0, (c), false, false)

// ---------------------------------------------------------------------------
// Pack a row-major f32 weight matrix W[N, Kd] into WMMA-B-operand tile order:
// tiles (n_tile, k_tile) of 16x32, 512 bf16 each; lane l of a tile holds
// W[n_tile*16 + (l&15), k_tile*32 + (l>>4)*16 + 0..15] contiguously (32 B).
// ---------------------------------------------------------------------------
__global__ void pack_w_f32(const float* __restrict__ src, bf16_t* __restrict__ dst,
                           int N, int Kd) {
  long total = (long)N * Kd;
  long p = (long)blockIdx.x * blockDim.x + threadIdx.x;
  if (p >= total) return;
  long tile = p >> 9;
  int  rem  = (int)(p & 511);
  int  lane = rem >> 4, i = rem & 15;
  int  kc   = Kd >> 5;
  long tn = tile / kc, tk = tile - tn * kc;
  int  n  = (int)tn * 16 + (lane & 15);
  int  k  = (int)tk * 32 + ((lane >> 4) << 4) + i;
  dst[p] = f2bf(src[(long)n * Kd + k]);
}

// Same packing from a strided bf16 source (used for k and v^T), batched over z.
__global__ void pack_b_bf16(const bf16_t* __restrict__ src, bf16_t* __restrict__ dst,
                            int N, int Kd, long sn, long sk,
                            long srcBatchStride, long dstBatchStride) {
  int bz = blockIdx.z;
  long total = (long)N * Kd;
  long p = (long)blockIdx.x * blockDim.x + threadIdx.x;
  if (p >= total) return;
  long tile = p >> 9;
  int  rem  = (int)(p & 511);
  int  lane = rem >> 4, i = rem & 15;
  int  kc   = Kd >> 5;
  long tn = tile / kc, tk = tile - tn * kc;
  int  n  = (int)tn * 16 + (lane & 15);
  int  k  = (int)tk * 32 + ((lane >> 4) << 4) + i;
  dst[(long)bz * dstBatchStride + p] =
      src[(long)bz * srcBatchStride + (long)n * sn + (long)k * sk];
}

// ---------------------------------------------------------------------------
// Embedding gather + f32->bf16 conversion: xbf[b,t,:] = bf16(emb[tokens[b,t],:])
// ---------------------------------------------------------------------------
__global__ void embed_gather(const int* __restrict__ tokens,
                             const float* __restrict__ emb,
                             bf16_t* __restrict__ xbf) {
  long total = (long)Bsz * Tsz * Esz;
  long p = (long)blockIdx.x * blockDim.x + threadIdx.x;
  if (p >= total) return;
  long bt = p / Esz;
  int  e  = (int)(p - bt * Esz);
  int  tok = tokens[bt];
  xbf[p] = f2bf(emb[(long)tok * Esz + e]);
}

// ---------------------------------------------------------------------------
// Generic batched WMMA GEMM:  out = alpha * (A[M,Kd] * Bpacked[N,Kd]^T + bias)
//
// Block = 8 waves covering one 64-row M-group x 8 consecutive 16-col N-tiles.
// K is processed in ROUNDS of 128 (4 chunks of 32):
//   * the 64x128 A panel is staged ONCE per block per round into LDS with 4
//     global_load_async_to_lds_b128 per thread (double-buffered 2x16KB;
//     s_wait_asynccnt 4 relies on in-order async completion, then one
//     barrier) -> 8x less A traffic and only 2 barriers per 16 WMMAs,
//   * each wave clause-loads its 4 private B fragments up front, consuming
//     them within the round so the compiler's drain-at-barrier costs nothing,
//   * the last round is peeled so the main loop body is branch-free.
// ---------------------------------------------------------------------------
__global__ void __launch_bounds__(256)
gemm_wmma_bf16(const bf16_t* __restrict__ A,  long Abs,
               const bf16_t* __restrict__ Bp, long Bbs,
               const float*  __restrict__ bias,
               float*  __restrict__ outF, long Fbs,
               bf16_t* __restrict__ outB, long Obs,
               int M, int N, int Kd, float alpha) {
  __shared__ alignas(64) bf16_t asmem[2][4 * 64 * 32];   // 2 x 16 KB A rounds

  const int bz = blockIdx.z;
  A  += (long)bz * Abs;
  Bp += (long)bz * Bbs;

  const int tid    = threadIdx.x;
  const int lane   = tid & 31;
  const int waveid = tid >> 5;
  const int ntb    = (N >> 4) >> 3;                  // blocks of 8 N-tiles
  const int g      = blockIdx.x / ntb;               // 64-row M-group
  const int tn     = (blockIdx.x - g * ntb) * 8 + waveid;
  const int kc     = Kd >> 5;                        // 32-wide chunks
  const int nr     = Kd >> 7;                        // 128-wide rounds

  // Async A staging: thread owns one 16B segment per chunk (64 rows x 64 B).
  const int arow = tid >> 2;
  const int aseg = tid & 3;
  const bf16_t*  agsrc  = A + (long)(g * 64 + arow) * Kd + aseg * 8;
  const uint32_t albase = (uint32_t)(uintptr_t)(&asmem[0][0]) + (uint32_t)(arow * 64 + aseg * 16);

  // WMMA fragment addressing
  const int mrow  = lane & 15;
  const int khalf = (lane >> 4) << 4;
  const bf16_t* afbase = &asmem[0][0] + mrow * 32 + khalf;

  const bf16_t* bbase = Bp + ((long)tn * kc) * 512 + lane * 16;

  v8f c0 = {0.f,0.f,0.f,0.f,0.f,0.f,0.f,0.f};
  v8f c1 = c0, c2 = c0, c3 = c0;

  // stage round r (4 chunks of 64x32) into LDS buffer (r&1)
  auto issue = [&](int r) {
    const bf16_t* gp = agsrc + (long)r * 128;
    uint32_t      lp = albase + (uint32_t)((r & 1) * 16384);
#pragma unroll
    for (int j = 0; j < 4; ++j)
      async_copy_b128(lp + (uint32_t)(j * 4096), gp + j * 32);
  };

  // compute one round: 4 B fragments, 16 ds A fragments, 16 WMMAs
  auto compute = [&](int r) {
    const bf16_t* bq = bbase + (long)(4 * r) * 512;
    v16bf barr[4];
#pragma unroll
    for (int j = 0; j < 4; ++j) barr[j] = *(const v16bf*)(bq + j * 512);
    __builtin_prefetch(bq + 4 * 512, 0, 0);            // next round's B
    const bf16_t* af = afbase + (r & 1) * (4 * 64 * 32);
#pragma unroll
    for (int j = 0; j < 4; ++j) {
      const bf16_t* afj = af + j * 2048;
      v16bf a;
      a  = *(const v16bf*)(afj + 0 * 512);  c0 = WMMA_BF16(a, barr[j], c0);
      a  = *(const v16bf*)(afj + 1 * 512);  c1 = WMMA_BF16(a, barr[j], c1);
      a  = *(const v16bf*)(afj + 2 * 512);  c2 = WMMA_BF16(a, barr[j], c2);
      a  = *(const v16bf*)(afj + 3 * 512);  c3 = WMMA_BF16(a, barr[j], c3);
    }
  };

  issue(0);
  for (int r = 0; r < nr - 1; ++r) {
    issue(r + 1);          // prefetch next round into the other LDS buffer
    wait_async_le4();      // in-order completion => round r has landed
    __syncthreads();       // round r visible to all 8 waves
    compute(r);
    __syncthreads();       // reads done; buffer (r&1) free for round r+2
  }
  wait_async_0();          // peeled last round
  __syncthreads();
  compute(nr - 1);

  // C/D layout: VGPR r, lanes 0-15 -> (M=r, N=lane); lanes 16-31 -> (M=8+r).
  const int   n  = (tn << 4) + (lane & 15);
  const float bv = bias ? bias[n] : 0.0f;
  const int   mb = (lane >> 4) << 3;
  v8f cc[4] = {c0, c1, c2, c3};
#pragma unroll
  for (int i = 0; i < 4; ++i) {
    const int mt = g * 64 + i * 16 + mb;
#pragma unroll
    for (int r = 0; r < 8; ++r) {
      float v = (cc[i][r] + bv) * alpha;
      long idx = (long)(mt + r) * N + n;
      if (outF) outF[(long)bz * Fbs + idx] = v;
      if (outB) outB[(long)bz * Obs + idx] = f2bf(v);
    }
  }
}

// ---------------------------------------------------------------------------
// Elman RNN scan: h_t = relu(xp[:,t,:] + h_{t-1} @ Wh^T + bh), serial over T.
// One 1024-thread workgroup (32 wave32 on one WGP). h is kept double-buffered
// in LDS as a zero-padded 16xH bf16 tile (rows 0..3 = batch). Each wave does
// 2 of the 64 output 16-col tiles per step via WMMA; Wh streams from L2 with
// a one-chunk B register double buffer (last chunk peeled: branch-free body).
// ---------------------------------------------------------------------------
__global__ void __launch_bounds__(1024)
rnn_scan(const float*  __restrict__ xp,     // [B,T,H] f32
         const bf16_t* __restrict__ pWh,    // packed [H,H]
         const float*  __restrict__ bh,
         bf16_t* __restrict__ hidbf,        // [B,T,H] bf16
         float*  __restrict__ hidf) {       // [B,T,H] f32 (d_out region)
  __shared__ alignas(64) bf16_t hbuf[2][16 * Hsz];   // 64 KB

  const int tid  = threadIdx.x;
  const int wave = tid >> 5, lane = tid & 31;
  bf16_t* flat = &hbuf[0][0];
  for (int i = tid; i < 2 * 16 * Hsz; i += 1024) flat[i] = f2bf(0.0f);
  __syncthreads();

  const int kc    = Hsz >> 5;          // 32 K-chunks
  const int mrow  = lane & 15;
  const int khalf = (lane >> 4) << 4;

  for (int t = 0; t < Tsz; ++t) {
    const bf16_t* cur = hbuf[t & 1];
    bf16_t*       nxt = hbuf[(t + 1) & 1];

    for (int half = 0; half < 2; ++half) {
      const int tn = wave + half * 32;   // 0..63 output tiles
      v8f c = {0.f,0.f,0.f,0.f,0.f,0.f,0.f,0.f};
      const bf16_t* ap = cur + mrow * Hsz + khalf;
      const bf16_t* bp = pWh + ((long)tn * kc) * 512 + lane * 16;
      v16bf bcur = *(const v16bf*)bp;
      for (int k = 0; k < kc - 1; ++k) {           // branch-free main body
        v16bf bnext = *(const v16bf*)(bp + (long)(k + 1) * 512);
        v16bf a = *(const v16bf*)(ap + k * 32);
        c = WMMA_BF16(a, bcur, c);
        bcur = bnext;
      }
      {                                            // peeled last chunk
        v16bf a = *(const v16bf*)(ap + (kc - 1) * 32);
        c = WMMA_BF16(a, bcur, c);
      }
      const int n = (tn << 4) + mrow;
      if (lane < 16) {                         // rows 0..7 live in lanes 0-15
        const float bb = bh[n];
#pragma unroll
        for (int r = 0; r < 4; ++r) {          // only batch rows 0..3 are real
          long idx = ((long)r * Tsz + t) * Hsz + n;
          float v = c[r] + xp[idx] + bb;
          v = v > 0.0f ? v : 0.0f;
          hidf[idx]  = v;
          hidbf[idx] = f2bf(v);
          nxt[r * Hsz + n] = f2bf(v);
        }
      }
    }
    __syncthreads();
  }
}

// ---------------------------------------------------------------------------
// Causal softmax over score rows; writes bf16 probs zero-padded beyond t so
// the following attn GEMM can run full K=T with no masking.
// ---------------------------------------------------------------------------
__global__ void __launch_bounds__(256)
softmax_causal(const float* __restrict__ scores, bf16_t* __restrict__ probs) {
  const int row = blockIdx.x;            // b*T + t
  const int t   = row & (Tsz - 1);
  const float* s = scores + (long)row * Tsz;
  bf16_t*      p = probs  + (long)row * Tsz;
  __shared__ float red[256];
  const int tid = threadIdx.x;

  float mx = -1e30f;
  for (int i = tid; i <= t; i += 256) mx = fmaxf(mx, s[i]);
  red[tid] = mx; __syncthreads();
  for (int o = 128; o > 0; o >>= 1) { if (tid < o) red[tid] = fmaxf(red[tid], red[tid + o]); __syncthreads(); }
  mx = red[0]; __syncthreads();

  float sum = 0.0f;
  for (int i = tid; i <= t; i += 256) sum += __expf(s[i] - mx);
  red[tid] = sum; __syncthreads();
  for (int o = 128; o > 0; o >>= 1) { if (tid < o) red[tid] += red[tid + o]; __syncthreads(); }
  const float inv = 1.0f / red[0];

  for (int i = tid; i < Tsz; i += 256)
    p[i] = (i <= t) ? f2bf(__expf(s[i] - mx) * inv) : f2bf(0.0f);
}

// ---------------------------------------------------------------------------
// Host-side orchestration
// ---------------------------------------------------------------------------
extern "C" void kernel_launch(void* const* d_in, const int* in_sizes, int n_in,
                              void* d_out, int out_size, void* d_ws, size_t ws_size,
                              hipStream_t stream) {
  (void)in_sizes; (void)n_in; (void)out_size; (void)ws_size;

  const int*   tokens = (const int*)  d_in[0];
  const float* emb    = (const float*)d_in[1];
  const float* Wi     = (const float*)d_in[2];
  const float* bi     = (const float*)d_in[3];
  const float* Wh     = (const float*)d_in[4];
  const float* bh     = (const float*)d_in[5];
  const float* Wro    = (const float*)d_in[6];
  const float* bro    = (const float*)d_in[7];
  const float* Wq     = (const float*)d_in[8];
  const float* bq     = (const float*)d_in[9];
  const float* Wk     = (const float*)d_in[10];
  const float* bk     = (const float*)d_in[11];
  const float* Wv     = (const float*)d_in[12];
  const float* bv     = (const float*)d_in[13];
  const float* Wao    = (const float*)d_in[14];
  const float* bao    = (const float*)d_in[15];
  const float* Wlm    = (const float*)d_in[16];
  const float* blm    = (const float*)d_in[17];

  // d_out = logits [B,T,V] ++ hidden [B,T,H] ++ outs [B,T,H], all f32
  float* logits = (float*)d_out;
  float* hidf   = logits + (long)Bsz * Tsz * Vsz;
  float* outsF  = hidf   + (long)Bsz * Tsz * Hsz;

  // workspace bump allocator (small pad on each buffer tolerates prefetch-ahead)
  char*  ws  = (char*)d_ws;
  size_t off = 0;
  auto alloc = [&](size_t bytes) -> void* {
    off = (off + 255) & ~(size_t)255;
    void* p = ws + off;
    off += bytes + 4096;
    return p;
  };

  bf16_t* xbf    = (bf16_t*)alloc((size_t)Bsz*Tsz*Esz*2);
  bf16_t* pWi    = (bf16_t*)alloc((size_t)Hsz*Esz*2);
  bf16_t* pWh    = (bf16_t*)alloc((size_t)Hsz*Hsz*2);
  bf16_t* pWro   = (bf16_t*)alloc((size_t)Hsz*Hsz*2);
  bf16_t* pWq    = (bf16_t*)alloc((size_t)Ksz*Hsz*2);
  bf16_t* pWk    = (bf16_t*)alloc((size_t)Ksz*Hsz*2);
  bf16_t* pWv    = (bf16_t*)alloc((size_t)Vdz*Hsz*2);
  bf16_t* pWao   = (bf16_t*)alloc((size_t)Hsz*Vdz*2);
  bf16_t* pWlm   = (bf16_t*)alloc((size_t)Vsz*Hsz*2);
  float*  xp     = (float*) alloc((size_t)Bsz*Tsz*Hsz*4);
  bf16_t* hidbf  = (bf16_t*)alloc((size_t)Bsz*Tsz*Hsz*2);
  bf16_t* outsbf = (bf16_t*)alloc((size_t)Bsz*Tsz*Hsz*2);
  bf16_t* qbf    = (bf16_t*)alloc((size_t)Bsz*Tsz*Ksz*2);
  bf16_t* kbf    = (bf16_t*)alloc((size_t)Bsz*Tsz*Ksz*2);
  bf16_t* vbf    = (bf16_t*)alloc((size_t)Bsz*Tsz*Vdz*2);
  bf16_t* pKb    = (bf16_t*)alloc((size_t)Bsz*Tsz*Ksz*2);
  bf16_t* pVT    = (bf16_t*)alloc((size_t)Bsz*Vdz*Tsz*2);
  float*  scores = (float*) alloc((size_t)Bsz*Tsz*Tsz*4);
  bf16_t* probs  = (bf16_t*)alloc((size_t)Bsz*Tsz*Tsz*2);
  bf16_t* attnbf = (bf16_t*)alloc((size_t)Bsz*Tsz*Vdz*2);
  bf16_t* aobf   = (bf16_t*)alloc((size_t)Bsz*Tsz*Hsz*2);

  auto blocks = [](long n) { return (unsigned)((n + 255) / 256); };

  // 1. pack all weights into WMMA-B tile order (f32 -> bf16)
  pack_w_f32<<<blocks((long)Hsz*Esz), 256, 0, stream>>>(Wi,  pWi,  Hsz, Esz);
  pack_w_f32<<<blocks((long)Hsz*Hsz), 256, 0, stream>>>(Wh,  pWh,  Hsz, Hsz);
  pack_w_f32<<<blocks((long)Hsz*Hsz), 256, 0, stream>>>(Wro, pWro, Hsz, Hsz);
  pack_w_f32<<<blocks((long)Ksz*Hsz), 256, 0, stream>>>(Wq,  pWq,  Ksz, Hsz);
  pack_w_f32<<<blocks((long)Ksz*Hsz), 256, 0, stream>>>(Wk,  pWk,  Ksz, Hsz);
  pack_w_f32<<<blocks((long)Vdz*Hsz), 256, 0, stream>>>(Wv,  pWv,  Vdz, Hsz);
  pack_w_f32<<<blocks((long)Hsz*Vdz), 256, 0, stream>>>(Wao, pWao, Hsz, Vdz);
  pack_w_f32<<<blocks((long)Vsz*Hsz), 256, 0, stream>>>(Wlm, pWlm, Vsz, Hsz);

  // 2. embedding gather
  embed_gather<<<blocks((long)Bsz*Tsz*Esz), 256, 0, stream>>>(tokens, emb, xbf);

  // GEMM launch helper: block = 64 rows x 128 cols (8 waves)
  auto gemm = [&](const bf16_t* A, long Abs, const bf16_t* Bp, long Bbs,
                  const float* bias, float* oF, long Fbs, bf16_t* oB, long Obs,
                  int M, int N, int Kd, float alpha, int batch) {
    unsigned nblk = (unsigned)((M >> 6) * ((N >> 4) >> 3));
    dim3 grid(nblk, 1, (unsigned)batch);
    gemm_wmma_bf16<<<grid, 256, 0, stream>>>(A, Abs, Bp, Bbs, bias,
                                             oF, Fbs, oB, Obs, M, N, Kd, alpha);
  };

  // 3. xp = x @ Wi^T + bi   (f32, consumed by RNN scan)
  gemm(xbf, 0, pWi, 0, bi, xp, 0, nullptr, 0, Bsz*Tsz, Hsz, Esz, 1.0f, 1);

  // 4. serial RNN recurrence (one workgroup, LDS double-buffered state)
  rnn_scan<<<1, 1024, 0, stream>>>(xp, pWh, bh, hidbf, hidf);

  // 5. outs = hidden @ Wro^T + bro   (f32 to d_out + bf16 for q/k/v)
  gemm(hidbf, 0, pWro, 0, bro, outsF, 0, outsbf, 0, Bsz*Tsz, Hsz, Hsz, 1.0f, 1);

  // 6. q/k/v projections (q pre-scaled by 1/sqrt(K))
  const float qscale = 1.0f / sqrtf((float)Ksz);
  gemm(outsbf, 0, pWq, 0, bq, nullptr, 0, qbf, 0, Bsz*Tsz, Ksz, Hsz, qscale, 1);
  gemm(outsbf, 0, pWk, 0, bk, nullptr, 0, kbf, 0, Bsz*Tsz, Ksz, Hsz, 1.0f, 1);
  gemm(outsbf, 0, pWv, 0, bv, nullptr, 0, vbf, 0, Bsz*Tsz, Vdz, Hsz, 1.0f, 1);

  // 7. repack k (as [T,K] "weights") and v^T (as [Vd,T]) per batch
  {
    dim3 g(blocks((long)Tsz*Ksz), 1, Bsz);
    pack_b_bf16<<<g, 256, 0, stream>>>(kbf, pKb, Tsz, Ksz, (long)Ksz, 1,
                                       (long)Tsz*Ksz, (long)Tsz*Ksz);
  }
  {
    dim3 g(blocks((long)Vdz*Tsz), 1, Bsz);
    pack_b_bf16<<<g, 256, 0, stream>>>(vbf, pVT, Vdz, Tsz, 1, (long)Vdz,
                                       (long)Tsz*Vdz, (long)Vdz*Tsz);
  }

  // 8. scores = q @ k^T   (batched, f32)
  gemm(qbf, (long)Tsz*Ksz, pKb, (long)Tsz*Ksz, nullptr,
       scores, (long)Tsz*Tsz, nullptr, 0, Tsz, Tsz, Ksz, 1.0f, Bsz);

  // 9. causal softmax -> zero-padded bf16 probs
  softmax_causal<<<Bsz*Tsz, 256, 0, stream>>>(scores, probs);

  // 10. attn = probs @ v   (batched, K = T, padding handles causality)
  gemm(probs, (long)Tsz*Tsz, pVT, (long)Vdz*Tsz, nullptr,
       nullptr, 0, attnbf, (long)Tsz*Vdz, Tsz, Vdz, Tsz, 1.0f, Bsz);

  // 11. attn_out = attn @ Wao^T + bao
  gemm(attnbf, 0, pWao, 0, bao, nullptr, 0, aobf, 0, Bsz*Tsz, Hsz, Vdz, 1.0f, 1);

  // 12. logits = attn_out @ Wlm^T + blm  -> d_out (the 268-GFLOP hot GEMM)
  gemm(aobf, 0, pWlm, 0, blm, logits, 0, nullptr, 0, Bsz*Tsz, Vsz, Hsz, 1.0f, 1);
}
